// ClusterMemory_26826365731329
// MI455X (gfx1250) — compile-verified, matrix operand order checked by reading, not run on hardware
//
#include <hip/hip_runtime.h>
#include <math.h>

#define BB 256
#define DD 2048
#define NN 30000
#define NT (NN / 16)          // 1875 column tiles (exact)
#define KC 64                 // K-chunk staged in LDS per barrier round
#define NCH (DD / KC)         // 32 chunks
#define INV_TEMP 20.0f        // 1/0.05
#define SHIFT 20.0f           // logits bounded by 1/temp (unit vectors)
#define EPSV 1e-12f
#define MOM 0.2f

typedef __attribute__((ext_vector_type(8)))  float  v8f;
typedef __attribute__((ext_vector_type(16))) __bf16 v16bf;
typedef __attribute__((ext_vector_type(8)))  __bf16 v8bf;

__device__ __forceinline__ unsigned short f2bfbits(float f) {
  return (unsigned short)(__float_as_uint(f) >> 16);   // truncating split: lo term captures remainder
}
__device__ __forceinline__ float bfbits2f(unsigned short h) {
  return __uint_as_float(((unsigned)h) << 16);
}

// ---------------- Kernel 1: row-normalize inputs, emit fp32 + bf16 hi/lo split ----------------
__global__ __launch_bounds__(256) void k_normalize(const float* __restrict__ in,
                                                   float* __restrict__ xn,
                                                   unsigned short* __restrict__ xhi,
                                                   unsigned short* __restrict__ xlo) {
  __shared__ float red[256];
  const int row = blockIdx.x;
  const int tid = threadIdx.x;
  const float* src = in + (size_t)row * DD;
  float ss = 0.f;
  for (int k = tid; k < DD; k += 256) { float v = src[k]; ss += v * v; }
  red[tid] = ss; __syncthreads();
  for (int s = 128; s > 0; s >>= 1) { if (tid < s) red[tid] += red[tid + s]; __syncthreads(); }
  const float inv = 1.0f / (sqrtf(red[0]) + EPSV);
  for (int k = tid; k < DD; k += 256) {
    float v = src[k] * inv;
    unsigned short hb = f2bfbits(v);
    xn [(size_t)row * DD + k] = v;
    xhi[(size_t)row * DD + k] = hb;
    xlo[(size_t)row * DD + k] = f2bfbits(v - bfbits2f(hb));
  }
}

// ---------------- Kernel 2: split-bf16 WMMA GEMM, LDS-staged B, fused bank copy ---------------
__global__ __launch_bounds__(256) void k_gemm(const float* __restrict__ feats,
                                              const unsigned short* __restrict__ xhi,
                                              const unsigned short* __restrict__ xlo,
                                              const int* __restrict__ targets,
                                              float* __restrict__ partial,
                                              float* __restrict__ tgtlogit,
                                              float* __restrict__ out_feats) {
  // [buf][hi/lo][col][K+pad]; 144B col stride: 16B-aligned, bank-spreading. 9.2 KB total.
  __shared__ __align__(16) unsigned short ls[2][2][16][72];
  __shared__ int tg[BB];

  const int ntile = blockIdx.x;
  const int n0    = ntile * 16;
  const int tid   = threadIdx.x;
  const int wave  = tid >> 5;
  const int lane  = tid & 31;
  const int c16   = lane & 15;          // output column / B-operand column
  const int hi16  = (lane >> 4) & 1;    // lane half selects K sub-block and C row half
  const int klocal = hi16 ? 8 : 0;

  tg[tid] = targets[tid];               // BB == blockDim.x

  // Fused copy: this tile's 16 feature rows -> output bank (also warms L0/L2 for the GEMM).
  // out_feats = d_out+1 (4-byte aligned only) => b32 stores, b128 loads.
  {
    const float4* s4 = (const float4*)(feats + (size_t)n0 * DD);
    float* dst = out_feats + (size_t)n0 * DD;
    for (int i = tid; i < 16 * DD / 4; i += 256) {
      float4 v = s4[i];
      float* d = dst + 4 * i;
      d[0] = v.x; d[1] = v.y; d[2] = v.z; d[3] = v.w;
    }
  }

  const int rt0 = wave, rt1 = wave + 8;          // two 16-row tiles per wave
  v8f acc0 = {}; v8f acc1 = {};

  const size_t aRow0 = (size_t)(rt0 * 16 + c16) * DD;
  const size_t aRow1 = (size_t)(rt1 * 16 + c16) * DD;

  // Cooperative conversion assignment: 16 threads per column, 4 consecutive K each (coalesced).
  const int ccol = tid >> 4;            // 0..15
  const int ckk  = (tid & 15) * 4;      // 0..60
  const float* csrc = feats + (size_t)(n0 + ccol) * DD + ckk;

  auto convert_store = [&](int b, int kbase) {
    float4 f = *(const float4*)(csrc + kbase);
    unsigned u0 = __float_as_uint(f.x), u1 = __float_as_uint(f.y);
    unsigned u2 = __float_as_uint(f.z), u3 = __float_as_uint(f.w);
    unsigned ph0 = (u0 >> 16) | (u1 & 0xffff0000u);
    unsigned ph1 = (u2 >> 16) | (u3 & 0xffff0000u);
    float l0 = f.x - __uint_as_float(u0 & 0xffff0000u);
    float l1 = f.y - __uint_as_float(u1 & 0xffff0000u);
    float l2 = f.z - __uint_as_float(u2 & 0xffff0000u);
    float l3 = f.w - __uint_as_float(u3 & 0xffff0000u);
    unsigned pl0 = (__float_as_uint(l0) >> 16) | (__float_as_uint(l1) & 0xffff0000u);
    unsigned pl1 = (__float_as_uint(l2) >> 16) | (__float_as_uint(l3) & 0xffff0000u);
    *(uint2*)&ls[b][0][ccol][ckk] = make_uint2(ph0, ph1);
    *(uint2*)&ls[b][1][ccol][ckk] = make_uint2(pl0, pl1);
  };

  convert_store(0, 0);                  // stage chunk 0
  __syncthreads();

  for (int ch = 0; ch < NCH; ++ch) {
    const int buf = ch & 1;
    if (ch + 2 < NCH) __builtin_prefetch(csrc + (ch + 2) * KC, 0, 0);
    if (ch + 1 < NCH) convert_store(buf ^ 1, (ch + 1) * KC);   // fill other buffer

#pragma unroll
    for (int s = 0; s < 2; ++s) {       // two 32-K WMMA steps per chunk
      const int kb_rel = s * 32 + klocal;
      // ---- B fragments from LDS (converted once per workgroup) ----
      const v8bf* bh = (const v8bf*)&ls[buf][0][c16][kb_rel];
      const v8bf* bl = (const v8bf*)&ls[buf][1][c16][kb_rel];
      v16bf bhi = __builtin_shufflevector(bh[0], bh[2], 0,1,2,3,4,5,6,7,8,9,10,11,12,13,14,15);
      v16bf blo = __builtin_shufflevector(bl[0], bl[2], 0,1,2,3,4,5,6,7,8,9,10,11,12,13,14,15);
      // ---- A fragments: precomputed bf16 hi/lo of normalized inputs (b128 global loads) ----
      const int kb = ch * KC + kb_rel;
      const v8bf* ph0 = (const v8bf*)(xhi + aRow0 + kb);
      const v8bf* pl0 = (const v8bf*)(xlo + aRow0 + kb);
      const v8bf* ph1 = (const v8bf*)(xhi + aRow1 + kb);
      const v8bf* pl1 = (const v8bf*)(xlo + aRow1 + kb);
      v16bf a0h = __builtin_shufflevector(ph0[0], ph0[2], 0,1,2,3,4,5,6,7,8,9,10,11,12,13,14,15);
      v16bf a0l = __builtin_shufflevector(pl0[0], pl0[2], 0,1,2,3,4,5,6,7,8,9,10,11,12,13,14,15);
      v16bf a1h = __builtin_shufflevector(ph1[0], ph1[2], 0,1,2,3,4,5,6,7,8,9,10,11,12,13,14,15);
      v16bf a1l = __builtin_shufflevector(pl1[0], pl1[2], 0,1,2,3,4,5,6,7,8,9,10,11,12,13,14,15);

      // D ~= Ah*Bh + Ah*Bl + Al*Bh  (fp32-grade accuracy from bf16 pipes)
      acc0 = __builtin_amdgcn_wmma_f32_16x16x32_bf16(false, a0h, false, bhi, (short)0, acc0, false, false);
      acc0 = __builtin_amdgcn_wmma_f32_16x16x32_bf16(false, a0h, false, blo, (short)0, acc0, false, false);
      acc0 = __builtin_amdgcn_wmma_f32_16x16x32_bf16(false, a0l, false, bhi, (short)0, acc0, false, false);
      acc1 = __builtin_amdgcn_wmma_f32_16x16x32_bf16(false, a1h, false, bhi, (short)0, acc1, false, false);
      acc1 = __builtin_amdgcn_wmma_f32_16x16x32_bf16(false, a1h, false, blo, (short)0, acc1, false, false);
      acc1 = __builtin_amdgcn_wmma_f32_16x16x32_bf16(false, a1l, false, bhi, (short)0, acc1, false, false);
    }
    __syncthreads();                    // one barrier per chunk (double-buffered)
  }

  // ---- epilogue: logits -> target gather + deterministic exp partials (shift = 20, no max pass)
  const int nb = n0 + c16;
#pragma unroll
  for (int i = 0; i < 8; ++i) {
    const int r0 = rt0 * 16 + i + 8 * hi16;
    const int r1 = rt1 * 16 + i + 8 * hi16;
    float l0 = acc0[i] * INV_TEMP;
    float l1 = acc1[i] * INV_TEMP;
    if (tg[r0] == nb) tgtlogit[r0] = l0;
    if (tg[r1] == nb) tgtlogit[r1] = l1;
    float e0 = expf(l0 - SHIFT);
    float e1 = expf(l1 - SHIFT);
#pragma unroll
    for (int m = 1; m <= 8; m <<= 1) {
      e0 += __shfl_xor(e0, m, 16);
      e1 += __shfl_xor(e1, m, 16);
    }
    if (c16 == 0) {
      partial[(size_t)r0 * NT + ntile] = e0;
      partial[(size_t)r1 * NT + ntile] = e1;
    }
  }
}

// ---------------- Kernel 3: deterministic per-row reduction of exp partials -------------------
__global__ __launch_bounds__(256) void k_rowsum(const float* __restrict__ partial,
                                                float* __restrict__ rowsum) {
  __shared__ float red[256];
  const int b = blockIdx.x, tid = threadIdx.x;
  float s = 0.f;
  for (int j = tid; j < NT; j += 256) s += partial[(size_t)b * NT + j];
  red[tid] = s; __syncthreads();
  for (int st = 128; st > 0; st >>= 1) { if (tid < st) red[tid] += red[tid + st]; __syncthreads(); }
  if (tid == 0) rowsum[b] = red[0];
}

// ---------------- Kernel 4: loss = mean(shift + log(sumexp) - target_logit) -------------------
__global__ __launch_bounds__(256) void k_loss(const float* __restrict__ rowsum,
                                              const float* __restrict__ tgtlogit,
                                              float* __restrict__ out) {
  __shared__ float red[256];
  const int tid = threadIdx.x;
  red[tid] = SHIFT + logf(rowsum[tid]) - tgtlogit[tid];
  __syncthreads();
  for (int st = 128; st > 0; st >>= 1) { if (tid < st) red[tid] += red[tid + st]; __syncthreads(); }
  if (tid == 0) out[0] = red[0] * (1.0f / (float)BB);
}

// ---------------- Kernel 5: sequential momentum update + renorm (scan semantics) --------------
__global__ __launch_bounds__(1024) void k_update(const float* __restrict__ xn,
                                                 const int* __restrict__ targets,
                                                 float* __restrict__ bank) {
  __shared__ float red[1024];
  const int tid = threadIdx.x;
  for (int i = 0; i < BB; ++i) {
    const int y = targets[i];
    float* f = bank + (size_t)y * DD;
    const float* x = xn + (size_t)i * DD;
    float v0 = MOM * f[tid]        + (1.0f - MOM) * x[tid];
    float v1 = MOM * f[tid + 1024] + (1.0f - MOM) * x[tid + 1024];
    red[tid] = v0 * v0 + v1 * v1;
    __syncthreads();
    for (int s = 512; s > 0; s >>= 1) { if (tid < s) red[tid] += red[tid + s]; __syncthreads(); }
    const float inv = 1.0f / (sqrtf(red[0]) + EPSV);
    f[tid]        = v0 * inv;
    f[tid + 1024] = v1 * inv;
    __syncthreads();   // protect red[] and written row before next scan step
  }
}

extern "C" void kernel_launch(void* const* d_in, const int* in_sizes, int n_in,
                              void* d_out, int out_size, void* d_ws, size_t ws_size,
                              hipStream_t stream) {
  (void)in_sizes; (void)n_in; (void)out_size; (void)ws_size;
  const float* inputs  = (const float*)d_in[0];
  const float* feats   = (const float*)d_in[1];
  const int*   targets = (const int*)d_in[2];
  float* out = (float*)d_out;            // [0] = loss, [1..] = new_features (30000x2048)

  // workspace layout (floats): xn | xhi(bf16) | xlo(bf16) | partial | tgtlogit | rowsum  (~6.1 MB)
  float* ws = (float*)d_ws;
  float*          xn  = ws;                                       // BB*DD f32
  unsigned short* xhi = (unsigned short*)(ws + (size_t)BB * DD);  // BB*DD u16
  unsigned short* xlo = xhi + (size_t)BB * DD;                    // BB*DD u16
  float* partial  = ws + 2 * (size_t)BB * DD;                     // BB*NT f32
  float* tgtl     = partial + (size_t)BB * NT;                    // BB
  float* rowsum   = tgtl + BB;                                    // BB

  k_normalize<<<BB, 256, 0, stream>>>(inputs, xn, xhi, xlo);
  k_gemm    <<<NT, 256, 0, stream>>>(feats, xhi, xlo, targets, partial, tgtl, out + 1);
  k_rowsum  <<<BB, 256, 0, stream>>>(partial, rowsum);
  k_loss    <<<1, 256, 0, stream>>>(rowsum, tgtl, out);
  k_update  <<<1, 1024, 0, stream>>>(xn, targets, out + 1);
}
